// ASIC_36421322670625
// MI455X (gfx1250) — matrix operand order repaired and checked by reading, not compiled, not run on hardware
//
#include <hip/hip_runtime.h>
#include <stdint.h>

// CDNA5 wave32 WMMA types
typedef __attribute__((ext_vector_type(16))) _Float16 v16h;
typedef __attribute__((ext_vector_type(8)))  float    v8f;

#define NPOS 512
#define PS   520   // padded f16 row stride (bank-conflict avoidance)

// Build an 8-entry soft-LUT row table: R[q] = f(x0,q>>2)*f(x1,(q>>1)&1)*f(x2,q&1),
// f(v,bit) = bit ? v : 1-v.   (constant indexing only -> stays in VGPRs)
#define MKR(R, x0, x1, x2) do {                                              \
    float a1=(x0), a0=1.0f-a1, b1=(x1), b0=1.0f-b1, c1=(x2), c0=1.0f-c1;     \
    float ab00=a0*b0, ab01=a0*b1, ab10=a1*b0, ab11=a1*b1;                    \
    R[0]=ab00*c0; R[1]=ab00*c1; R[2]=ab01*c0; R[3]=ab01*c1;                  \
    R[4]=ab10*c0; R[5]=ab10*c1; R[6]=ab11*c0; R[7]=ab11*c1; } while (0)

// One layer. Grid: 64 blocks (one per 16-pixel half-row), 512 threads (16 waves).
// Wave w == batch b; lanes%16 == pixel-in-group.
__global__ __launch_bounds__(512) void asic_layer(const float* __restrict__ in,
                                                  const float* __restrict__ twl,
                                                  float* __restrict__ out)
{
    __shared__ __align__(16) float     ldsA[NPOS * 16]; // [p][j] raw gates (32KB)
    __shared__ __align__(8)  _Float16  ldsB[16 * PS];   // [pixel][p] sigmoid f16

    const int tid   = threadIdx.x;
    const int lane  = tid & 31;
    const int b     = tid >> 5;              // wave index == batch
    const int h     = blockIdx.x >> 1;       // row of this pixel group
    const int wbase = (blockIdx.x & 1) << 4; // first column of the group

    // ---- Phase 1: async-stage the 512x16 f32 gate tile into LDS ------------
    // tg layout (L,NPOS,H,W): row p of this group = 16 consecutive floats.
    // 2048 x 16B chunks, 4 per thread, via GLOBAL_LOAD_ASYNC_TO_LDS_B128.
    const uint32_t lds0 = (uint32_t)(uintptr_t)&ldsA[0];
#pragma unroll
    for (int r = 0; r < 4; ++r) {
        int c = tid + (r << 9);                  // chunk id 0..2047
        int p = c >> 2, q = c & 3;
        uint64_t ga = (uint64_t)(uintptr_t)(twl + (p << 10) + (h << 5) + wbase + (q << 2));
        uint32_t la = lds0 + ((uint32_t)c << 4);
        asm volatile("global_load_async_to_lds_b128 %0, %1, off"
                     :: "v"(la), "v"(ga) : "memory");
    }
    asm volatile("s_wait_asynccnt 0" ::: "memory");
    __syncthreads();

    // ---- Per-lane soft-LUT factor tables for pixel = lane%16 ---------------
    // Window: rows h+0,h+1,h+2 ; cols w-1,w,w+1 (wrap), bit m = p>>(8-m).
    const int pix = lane & 15;
    const int wc  = wbase + pix;
    float R0[8], R1[8], R2[8];
    {
        float vv[9];
#pragma unroll
        for (int i = 0; i < 3; ++i) {
            int hr = (h + i) & 31;
#pragma unroll
            for (int j = 0; j < 3; ++j) {
                int wcj = (wc + j + 31) & 31;    // column offsets -1,0,+1
                vv[i * 3 + j] = in[(b << 10) + (hr << 5) + wcj];
            }
        }
        MKR(R0, vv[0], vv[1], vv[2]);
        MKR(R1, vv[3], vv[4], vv[5]);
        MKR(R2, vv[6], vv[7], vv[8]);
    }

    // ---- Phase 2: sigmoid once per gate (shared by all 16 batches) ---------
#pragma unroll
    for (int k = 0; k < 16; ++k) {
        int e = tid + (k << 9);                  // 0..8191
        float g = ldsA[e];
        float s = 1.0f / (1.0f + __expf(-g));
        int p = e >> 4, j = e & 15;
        ldsB[j * PS + p] = (_Float16)s;
    }
    __syncthreads();

    // ---- Phase 3: D = W(16pix x 512) * TW(512 x 16pix), diag(D) = outputs --
    // A layout (16-bit A 16x32, ISA 7.12.2): lanes 0-15 hold K={0..7,16..23},
    // lanes 16-31 hold K={8..15,24..31}; half t -> K = (t<8?t:t+8) + (hi?8:0).
    // B layout: lane n = column n (pixel), half t -> K = (t<8?t:t+8) + (hi?8:0).
    v8f acc = {};
    const bool hi = (lane >= 16);
    const int  pixbase = pix * PS;
#pragma unroll
    for (int c = 0; c < 16; ++c) {
        v16h a, bm;
#pragma unroll
        for (int t = 0; t < 16; ++t) {
            const int Klo = (t < 8) ? t : (t + 8);
            const int plo = (c << 5) + Klo;      // p for lanes 0-15
            const int phi = plo + 8;             // p for lanes 16-31
            float wl = R0[(plo >> 6) & 7] * R1[(plo >> 3) & 7] * R2[plo & 7];
            float wh = R0[(phi >> 6) & 7] * R1[(phi >> 3) & 7] * R2[phi & 7];
            a[t] = (_Float16)(hi ? wh : wl);
            int pb = (c << 5) + Klo + (hi ? 8 : 0);
            bm[t] = ldsB[pixbase + pb];
        }
        // v_wmma_f32_16x16x32_f16, f32 accumulate
        acc = __builtin_amdgcn_wmma_f32_16x16x32_f16(
                  false, a, false, bm, (short)0, acc, false, false);
    }

    // ---- Diagonal extract + clip + store -----------------------------------
    // D element (d,d): VGPR d&7, lane d (d<8) or lane d+16 (d>=8).
    int d = (lane < 8) ? lane : ((lane >= 24) ? (lane - 16) : -1);
    if (d >= 0) {
        float val = 0.0f;
#pragma unroll
        for (int g = 0; g < 8; ++g)
            if ((lane & 7) == g) val = acc[g];
        val = fminf(fmaxf(val, 0.0f), 1.0f);
        out[(b << 10) + (h << 5) + wbase + d] = val;
    }
}

extern "C" void kernel_launch(void* const* d_in, const int* in_sizes, int n_in,
                              void* d_out, int out_size, void* d_ws, size_t ws_size,
                              hipStream_t stream)
{
    const float* x  = (const float*)d_in[0];                 // (16,32,32)
    const float* tg = (const float*)d_in[1];                 // (4,512,32,32)
    float* out = (float*)d_out;
    float* s0  = (float*)d_ws;                               // 64KB ping buffer
    const size_t LSTRIDE = (size_t)512 * 32 * 32;            // per-layer gates

    // layer chain: x -> s0 -> out -> s0 -> out
    asic_layer<<<64, 512, 0, stream>>>(x,   tg + 0 * LSTRIDE, s0);
    asic_layer<<<64, 512, 0, stream>>>(s0,  tg + 1 * LSTRIDE, out);
    asic_layer<<<64, 512, 0, stream>>>(out, tg + 2 * LSTRIDE, s0);
    asic_layer<<<64, 512, 0, stream>>>(s0,  tg + 3 * LSTRIDE, out);
}
